// DecisionMamba_90563680403742
// MI455X (gfx1250) — matrix operand order repaired and checked by reading, not compiled, not run on hardware
//
#include <hip/hip_runtime.h>

// ---------------------------------------------------------------------------
// DecisionMamba forward for MI455X (gfx1250, wave32, WMMA).
// Bandwidth-bound (~0.5 TFLOP vs ~GB of traffic): GEMMs use
// v_wmma_f32_16x16x32_f16 (f32 accumulate) with LDS tiles laid out so every
// WMMA fragment is exactly two aligned ds_load_b128s per lane; global staging
// is float4-vectorized. The serial selective scan is register-resident f32.
// ---------------------------------------------------------------------------

typedef __attribute__((ext_vector_type(16))) _Float16 v16h;
typedef __attribute__((ext_vector_type(8)))  _Float16 h8;
typedef __attribute__((ext_vector_type(8)))  float    v8f;

#define ROWS 16384   // B * 4 * T = 8 * 2048
#define HD   768
#define EDI  1536
#define LSEQ 2048
#define NLAY 4

__device__ __forceinline__ v16h cat16(h8 lo, h8 hi) {
    return __builtin_shufflevector(lo, hi, 0,1,2,3,4,5,6,7,8,9,10,11,12,13,14,15);
}

// ============================ generic WMMA GEMM ============================
// C[M,N] = op( (A .* rowScale .* colScale) @ W )  with optional bias/act.
// Block: 256 threads (8 waves). Tile: 128(M) x 64(N), K-step 32.
// act: 0 = store, 1 = softplus(store), 2 = accumulate (+=).
// Cols n <  splitN -> out0[m*ld0+n] ; n >= splitN -> out1[m*ld1+(n-splitN)].
__global__ __launch_bounds__(256)
void wmma_gemm_kernel(const float* __restrict__ A, int lda,
                      const float* __restrict__ rowScale,
                      const float* __restrict__ colScale,
                      const float* __restrict__ W, int ldw,
                      int M, int N, int K,
                      float* __restrict__ out0, int ld0,
                      float* __restrict__ out1, int ld1, int splitN,
                      const float* __restrict__ bias, int act)
{
    // pitch 40 halves = 80B: multiple of 16B (aligned b128 reads) and
    // 20 DWORDs -> 20*m mod 64 is a permutation over 16 lanes (no conflicts).
    __shared__ __align__(16) _Float16 sA [128][40];  // A tile, [m][k]  10 KB
    __shared__ __align__(16) _Float16 sWt[ 64][40];  // W tile, [n][k]   5 KB

    const int tid  = threadIdx.x;
    const int wave = tid >> 5;
    const int lane = tid & 31;
    const int ml   = lane & 15;   // M (A frag) / N (B frag) coordinate
    const int ks   = lane >> 4;   // K-half selector
    const int m0b  = blockIdx.y * 128;
    const int n0b  = blockIdx.x * 64;
    const int mrow = m0b + wave * 16;

    v8f acc[4] = {};

    for (int k0 = 0; k0 < K; k0 += 32) {
        if (k0 + 32 < K)   // stream next weight tile toward L2/L0
            __builtin_prefetch(&W[(size_t)(k0 + 32) * ldw + n0b + (tid & 63)], 0, 1);

        // --- stage A tile: 128x32 f32 -> f16, float4 loads (4 per thread)
        #pragma unroll
        for (int i = 0; i < 4; i++) {
            int idx = i * 256 + tid;            // 0..1023
            int r   = idx >> 3;                 // 0..127
            int c4  = (idx & 7) << 2;           // 0,4,...,28
            int gr  = m0b + r, gk = k0 + c4;
            float4 v = make_float4(0.f, 0.f, 0.f, 0.f);
            if (gr < M) {
                if (gk + 3 < K) {
                    v = *(const float4*)&A[(size_t)gr * lda + gk];
                } else {
                    const float* ar = &A[(size_t)gr * lda];
                    if (gk + 0 < K) v.x = ar[gk + 0];
                    if (gk + 1 < K) v.y = ar[gk + 1];
                    if (gk + 2 < K) v.z = ar[gk + 2];
                    if (gk + 3 < K) v.w = ar[gk + 3];
                }
                float s = rowScale ? rowScale[gr] : 1.f;
                v.x *= s; v.y *= s; v.z *= s; v.w *= s;
                if (colScale) {
                    if (gk + 0 < K) v.x *= colScale[gk + 0];
                    if (gk + 1 < K) v.y *= colScale[gk + 1];
                    if (gk + 2 < K) v.z *= colScale[gk + 2];
                    if (gk + 3 < K) v.w *= colScale[gk + 3];
                }
            }
            h8* dst = (h8*)&sA[r][c4 & 24];   // write 4 halves via scalar stores
            (void)dst;
            sA[r][c4 + 0] = (_Float16)v.x;
            sA[r][c4 + 1] = (_Float16)v.y;
            sA[r][c4 + 2] = (_Float16)v.z;
            sA[r][c4 + 3] = (_Float16)v.w;
        }
        // --- stage W tile transposed: 32x64 f32 -> sWt[n][k] (2 float4/thread)
        #pragma unroll
        for (int i = 0; i < 2; i++) {
            int idx = i * 256 + tid;            // 0..511
            int r   = idx >> 4;                 // k-local 0..31
            int c4  = (idx & 15) << 2;          // n-local 0,4,...,60
            int gk  = k0 + r, gn = n0b + c4;
            float4 v = make_float4(0.f, 0.f, 0.f, 0.f);
            if (gk < K) {
                if (gn + 3 < N) {
                    v = *(const float4*)&W[(size_t)gk * ldw + gn];
                } else {
                    const float* wr = &W[(size_t)gk * ldw];
                    if (gn + 0 < N) v.x = wr[gn + 0];
                    if (gn + 1 < N) v.y = wr[gn + 1];
                    if (gn + 2 < N) v.z = wr[gn + 2];
                    if (gn + 3 < N) v.w = wr[gn + 3];
                }
            }
            sWt[c4 + 0][r] = (_Float16)v.x;
            sWt[c4 + 1][r] = (_Float16)v.y;
            sWt[c4 + 2][r] = (_Float16)v.z;
            sWt[c4 + 3][r] = (_Float16)v.w;
        }
        __syncthreads();

        // --- A fragment: lane (ml,ks) = rows of K {8ks..8ks+7, 16+8ks..23+8ks}
        //     two aligned ds_load_b128
        h8 a_lo = *(const h8*)&sA[wave * 16 + ml][ks * 8];
        h8 a_hi = *(const h8*)&sA[wave * 16 + ml][16 + ks * 8];
        v16h af = cat16(a_lo, a_hi);

        // --- 4 B fragments: lane (n,ks) = K {8ks..8ks+7, 16+8ks..23+8ks}
        #pragma unroll
        for (int j = 0; j < 4; j++) {
            const _Float16* bp = &sWt[j * 16 + ml][8 * ks];
            h8 b_lo = *(const h8*)(bp);
            h8 b_hi = *(const h8*)(bp + 16);
            v16h bf = cat16(b_lo, b_hi);
            acc[j] = __builtin_amdgcn_wmma_f32_16x16x32_f16(
                false, af, false, bf, (short)0, acc[j], false, false);
        }
        __syncthreads();
    }

    // --- epilogue: C/D layout M = mrow + r + 8*ks, N = n0b + 16j + ml
    #pragma unroll
    for (int j = 0; j < 4; j++) {
        int n = n0b + j * 16 + ml;
        if (n >= N) continue;
        #pragma unroll
        for (int r = 0; r < 8; r++) {
            int m = mrow + r + 8 * ks;
            if (m >= M) continue;
            float c = acc[j][r];
            if (bias) c += bias[n];
            if (act == 1) c = (c > 20.f) ? c : log1pf(__expf(c));   // softplus
            if (n < splitN) {
                if (act == 2) out0[(size_t)m * ld0 + n] += c;
                else          out0[(size_t)m * ld0 + n]  = c;
            } else {
                out1[(size_t)m * ld1 + (n - splitN)] = c;
            }
        }
    }
}

// ======================= embedding + layernorm =============================
__global__ __launch_bounds__(256)
void embed_ln_kernel(const float* __restrict__ states, const int* __restrict__ done,
                     const float* __restrict__ rtg, const float* __restrict__ hr,
                     const float* __restrict__ W_state, const float* __restrict__ b_state,
                     const float* __restrict__ W_return, const float* __restrict__ b_return,
                     const float* __restrict__ W_reward, const float* __restrict__ b_reward,
                     const float* __restrict__ E_done,
                     const float* __restrict__ ln_g, const float* __restrict__ ln_b,
                     float* __restrict__ x)
{
    __shared__ float red[256];
    __shared__ float srow[64];
    const int row = blockIdx.x;           // 0..16383
    const int b = row >> 11, p = row & 2047;
    const int t = p >> 2, type = p & 3;
    const int bt = b * 512 + t;
    const int tid = threadIdx.x;

    if (type == 1 && tid < 64) srow[tid] = states[(size_t)bt * 64 + tid];
    __syncthreads();

    float v[3];
    #pragma unroll
    for (int i = 0; i < 3; i++) {
        int h = tid + i * 256;
        float val;
        if (type == 0)      val = rtg[bt] * W_return[h] + b_return[h];
        else if (type == 1) {
            float s = b_state[h];
            for (int k = 0; k < 64; k++) s += srow[k] * W_state[k * HD + h];
            val = s;
        }
        else if (type == 2) val = hr[bt] * W_reward[h] + b_reward[h];
        else                val = E_done[done[bt] * HD + h];
        v[i] = val;
    }
    float s = v[0] + v[1] + v[2];
    red[tid] = s; __syncthreads();
    for (int o = 128; o > 0; o >>= 1) { if (tid < o) red[tid] += red[tid + o]; __syncthreads(); }
    float mu = red[0] / (float)HD;
    __syncthreads();
    float sq = 0.f;
    #pragma unroll
    for (int i = 0; i < 3; i++) { float d = v[i] - mu; sq += d * d; }
    red[tid] = sq; __syncthreads();
    for (int o = 128; o > 0; o >>= 1) { if (tid < o) red[tid] += red[tid + o]; __syncthreads(); }
    float inv = rsqrtf(red[0] / (float)HD + 1e-5f);
    #pragma unroll
    for (int i = 0; i < 3; i++) {
        int h = tid + i * 256;
        x[(size_t)row * HD + h] = (v[i] - mu) * inv * ln_g[h] + ln_b[h];
    }
}

// ===================== per-row rmsnorm scale (wave-per-row) ================
__global__ __launch_bounds__(256)
void rms_scale_kernel(const float* __restrict__ x, float* __restrict__ rs, int M, int K)
{
    int wave = threadIdx.x >> 5, lane = threadIdx.x & 31;
    int row = blockIdx.x * 8 + wave;
    if (row >= M) return;
    const float* xr = x + (size_t)row * K;
    float s = 0.f;
    for (int k = lane; k < K; k += 32) { float v = xr[k]; s += v * v; }
    for (int o = 16; o > 0; o >>= 1) s += __shfl_xor(s, o, 32);
    if (lane == 0) rs[row] = rsqrtf(s / (float)K + 1e-5f);
}

// ===================== causal depthwise conv (k=4) + SiLU ==================
__global__ __launch_bounds__(256)
void conv_silu_kernel(const float* __restrict__ xin, const float* __restrict__ cw,
                      const float* __restrict__ cb, float* __restrict__ xc)
{
    int e = blockIdx.x * 256 + threadIdx.x;   // channel
    int row = blockIdx.y;                      // token row
    int t = row & (LSEQ - 1);
    float acc = cb[e];
    #pragma unroll
    for (int j = 0; j < 4; j++) {
        int tt = t - 3 + j;
        if (tt >= 0) acc += xin[(size_t)(row - 3 + j) * EDI + e] * cw[e * 4 + j];
    }
    xc[(size_t)row * EDI + e] = acc / (1.f + __expf(-acc));
}

// ======================== selective scan (serial in t) =====================
__global__ __launch_bounds__(256)
void ssm_scan_kernel(const float* __restrict__ delta, const float* __restrict__ dbc,
                     const float* __restrict__ xc, const float* __restrict__ res,
                     const float* __restrict__ A_log, const float* __restrict__ D_ssm,
                     float* __restrict__ y)
{
    int c = blockIdx.x * 256 + threadIdx.x;
    int b = blockIdx.y;
    float a[16], h[16];
    #pragma unroll
    for (int n = 0; n < 16; n++) { a[n] = -__expf(A_log[c * 16 + n]); h[n] = 0.f; }
    float D = D_ssm[c];
    for (int t = 0; t < LSEQ; t++) {
        size_t row = (size_t)b * LSEQ + t;
        float d  = delta[row * EDI + c];
        float xv = xc[row * EDI + c];
        const float* bc = dbc + row * 80 + 48;   // B[16] then C[16]
        float dx = d * xv, yv = 0.f;
        #pragma unroll
        for (int n = 0; n < 16; n++) {
            float dA = __expf(d * a[n]);         // v_exp_f32 (trans unit)
            h[n] = dA * h[n] + dx * bc[n];
            yv += h[n] * bc[16 + n];
        }
        yv += D * xv;
        float r = res[row * EDI + c];
        yv *= r / (1.f + __expf(-r));            // * silu(res)
        y[row * EDI + c] = yv;
    }
}

// ============== synth out_proj (reference defaults it internally) ==========
__global__ void wout_init_kernel(float* __restrict__ w, int n)
{
    int i = blockIdx.x * blockDim.x + threadIdx.x;
    if (i >= n) return;
    unsigned u = (unsigned)i * 2654435761u;
    u ^= u >> 16; u *= 2246822519u; u ^= u >> 13; u *= 3266489917u; u ^= u >> 16;
    float f = (float)u * (1.0f / 4294967296.0f);
    w[i] = (f - 0.5f) * 0.069282f;               // ~N(0, 0.02) scale
}

// ===================== final rmsnorm + select + z head =====================
__global__ __launch_bounds__(256)
void head_kernel(const float* __restrict__ x, const float* __restrict__ rs,
                 const float* __restrict__ nfw, const float* __restrict__ Wz,
                 const float* __restrict__ bz, float* __restrict__ out)
{
    __shared__ float xn[HD];
    int i = blockIdx.x;                 // 0..4095  (b*512 + t)
    int b = i >> 9, t = i & 511;
    size_t row = (size_t)b * LSEQ + 4 * t + 1;   // state-token rows
    int tid = threadIdx.x;
    float scale = rs[row];
    for (int h = tid; h < HD; h += 256) xn[h] = x[row * HD + h] * scale * nfw[h];
    __syncthreads();
    if (tid < 32) {
        float s = bz[tid];
        for (int k = 0; k < HD; k++) s += xn[k] * Wz[k * 32 + tid];
        out[(size_t)i * 32 + tid] = tanhf(s);
    }
}

// ================================ driver ===================================
extern "C" void kernel_launch(void* const* d_in, const int* in_sizes, int n_in,
                              void* d_out, int out_size, void* d_ws, size_t ws_size,
                              hipStream_t stream)
{
    (void)in_sizes; (void)n_in; (void)out_size; (void)ws_size;
    const float* states    = (const float*)d_in[0];
    const int*   done      = (const int*)  d_in[2];
    const float* rtg       = (const float*)d_in[3];
    const float* hr        = (const float*)d_in[4];
    const float* W_state   = (const float*)d_in[5];
    const float* b_state   = (const float*)d_in[6];
    const float* W_return  = (const float*)d_in[7];
    const float* b_return  = (const float*)d_in[8];
    const float* W_reward  = (const float*)d_in[9];
    const float* b_reward  = (const float*)d_in[10];
    const float* E_done    = (const float*)d_in[11];
    const float* ln_g      = (const float*)d_in[12];
    const float* ln_b      = (const float*)d_in[13];
    const float* in_proj_w = (const float*)d_in[14];
    const float* conv_w    = (const float*)d_in[15];
    const float* conv_b    = (const float*)d_in[16];
    const float* x_proj_w  = (const float*)d_in[17];
    const float* dt_proj_w = (const float*)d_in[18];
    const float* dt_proj_b = (const float*)d_in[19];
    const float* A_log     = (const float*)d_in[20];
    const float* D_ssm     = (const float*)d_in[21];
    const float* norm_w    = (const float*)d_in[22];
    const float* norm_f_w  = (const float*)d_in[23];
    const float* W_z       = (const float*)d_in[24];
    const float* b_z       = (const float*)d_in[25];

    char* p = (char*)d_ws;
    float* x     = (float*)p; p += (size_t)ROWS * HD  * 4;
    float* rs    = (float*)p; p += (size_t)ROWS       * 4;
    float* xin   = (float*)p; p += (size_t)ROWS * EDI * 4;   // reused as y
    float* res   = (float*)p; p += (size_t)ROWS * EDI * 4;
    float* xc    = (float*)p; p += (size_t)ROWS * EDI * 4;
    float* dbc   = (float*)p; p += (size_t)ROWS * 80  * 4;
    float* delta = (float*)p; p += (size_t)ROWS * EDI * 4;
    float* wout  = (float*)p; p += (size_t)NLAY * EDI * HD * 4;

    { int n = NLAY * EDI * HD;
      wout_init_kernel<<<(n + 255) / 256, 256, 0, stream>>>(wout, n); }

    embed_ln_kernel<<<ROWS, 256, 0, stream>>>(states, done, rtg, hr,
        W_state, b_state, W_return, b_return, W_reward, b_reward,
        E_done, ln_g, ln_b, x);

    for (int l = 0; l < NLAY; l++) {
        rms_scale_kernel<<<ROWS / 8, 256, 0, stream>>>(x, rs, ROWS, HD);

        // xz = rmsnorm(x) @ in_proj_w[l]   (split into xin | res)
        wmma_gemm_kernel<<<dim3(2 * EDI / 64, ROWS / 128), 256, 0, stream>>>(
            x, HD, rs, norm_w + (size_t)l * HD,
            in_proj_w + (size_t)l * HD * 2 * EDI, 2 * EDI,
            ROWS, 2 * EDI, HD, xin, EDI, res, EDI, EDI, nullptr, 0);

        conv_silu_kernel<<<dim3(EDI / 256, ROWS), 256, 0, stream>>>(
            xin, conv_w + (size_t)l * EDI * 4, conv_b + (size_t)l * EDI, xc);

        // dbc = xc @ x_proj_w[l]   (N=80: dt|B|C)
        wmma_gemm_kernel<<<dim3(2, ROWS / 128), 256, 0, stream>>>(
            xc, EDI, nullptr, nullptr,
            x_proj_w + (size_t)l * EDI * 80, 80,
            ROWS, 80, EDI, dbc, 80, nullptr, 0, 80, nullptr, 0);

        // delta = softplus(dbc[:, :48] @ dt_proj_w[l] + dt_b)
        wmma_gemm_kernel<<<dim3(EDI / 64, ROWS / 128), 256, 0, stream>>>(
            dbc, 80, nullptr, nullptr,
            dt_proj_w + (size_t)l * 48 * EDI, EDI,
            ROWS, EDI, 48, delta, EDI, nullptr, 0, EDI,
            dt_proj_b + (size_t)l * EDI, 1);

        // selective scan -> y (reuses xin buffer)
        ssm_scan_kernel<<<dim3(EDI / 256, 8), 256, 0, stream>>>(
            delta, dbc, xc, res,
            A_log + (size_t)l * EDI * 16, D_ssm + (size_t)l * EDI, xin);

        // x += y @ out_proj[l]
        wmma_gemm_kernel<<<dim3(HD / 64, ROWS / 128), 256, 0, stream>>>(
            xin, EDI, nullptr, nullptr,
            wout + (size_t)l * EDI * HD, HD,
            ROWS, HD, EDI, x, HD, nullptr, 0, HD, nullptr, 2);
    }

    rms_scale_kernel<<<ROWS / 8, 256, 0, stream>>>(x, rs, ROWS, HD);
    head_kernel<<<8 * 512, 256, 0, stream>>>(x, rs, norm_f_w, W_z, b_z,
                                             (float*)d_out);
}
